// CosClassifier_45561013075980
// MI455X (gfx1250) — compile-verified, matrix operand order checked by reading, not run on hardware
//
#include <hip/hip_runtime.h>
#include <math.h>

// ---------------- problem constants ----------------
#define JNT   15
#define DIM   128
#define ANGD  3
#define EDIM  1965          // JNT*DIM + JNT*ANGD
#define FEAT  1920          // JNT*DIM
#define NCLS  68
#define BROWS 16384
#define NTILES 5            // ceil(68/16)
#define TEMP_INV (1.0f/200.0f)
#define OUTSCALE (16.0f * 15.0f)   // SCALE * J

typedef __attribute__((ext_vector_type(16))) _Float16 v16h;
typedef __attribute__((ext_vector_type(8)))  float    v8f;

// ---------------- workspace layout (bytes) ----------------
#define RNORM_OFF 0                          // BROWS floats (65536 B)
#define WNORM_OFF (BROWS * 4)                // 68 floats, padded to 1024 B
#define BFRAG_OFF (BROWS * 4 + 1024)         // 300 fragments * 1024 B = 307200 B
// total workspace use: ~374 KB

// =====================================================================
// Kernel 1: per-row inverse L2 norm over the first FEAT elements.
// (Used only as a final output scale; the GEMM runs on raw values.)
// =====================================================================
__global__ void cc_rownorm_kernel(const float* __restrict__ x,
                                  float* __restrict__ rnorm) {
    __shared__ float red[128];
    const int row = blockIdx.x;
    const float* p = x + (size_t)row * EDIM;
    float s = 0.f;
    for (int i = threadIdx.x; i < FEAT; i += blockDim.x) {
        float v = p[i];
        s += v * v;
    }
    red[threadIdx.x] = s;
    __syncthreads();
    for (int off = blockDim.x >> 1; off > 0; off >>= 1) {
        if ((int)threadIdx.x < off) red[threadIdx.x] += red[threadIdx.x + off];
        __syncthreads();
    }
    if (threadIdx.x == 0)
        rnorm[row] = 1.0f / fmaxf(sqrtf(red[0]), 1e-12f);
}

// =====================================================================
// Kernel 2: convert RAW W features to f16 and pack into WMMA B-fragment
// layout (no normalization here — folded into the epilogue scale).
// Fragment id = ((tn*JNT + j)*4 + kb).  Lane L holds class row
// n = tn*16 + (L&15), group g = L>>4; its v16h covers
// K = [kb*32+g*8, +8) ++ [kb*32+16+g*8, +8).  Rows >= 68 are zeros.
// =====================================================================
__global__ void cc_packB_kernel(const float* __restrict__ W,
                                v16h* __restrict__ bfrag) {
    const int frag = blockIdx.x;      // 0..299
    const int lane = threadIdx.x;     // 0..31
    const int kb = frag & 3;
    const int j  = (frag >> 2) % JNT;
    const int tn = frag / (4 * JNT);
    const int t = lane & 15, g = lane >> 4;
    const int n = tn * 16 + t;

    v16h out;
    if (n < NCLS) {
        const float* p = W + (size_t)n * EDIM + j * DIM + kb * 32 + g * 8;
#pragma unroll
        for (int e = 0; e < 8; ++e) out[e]     = (_Float16)p[e];
#pragma unroll
        for (int e = 0; e < 8; ++e) out[8 + e] = (_Float16)p[16 + e];
    } else {
#pragma unroll
        for (int e = 0; e < 16; ++e) out[e] = (_Float16)0.f;
    }
    bfrag[(size_t)frag * 32 + lane] = out;
}

// =====================================================================
// Kernel 3: fused GEMM + angle-softmax epilogue on raw values.
// One wave computes one 16(M) x 16(N) output tile.  For each joint j:
//   c = 0; 4x v_wmma_f32_16x16x32_f16 over K=128 (raw f16 inputs);
//   e_j = exp(||x_ang - p_ang|| / 200) per (m,n) element;
//   accS += e_j * c;  accE += e_j;
// logit = 240 * rx[m] * rw[n] * accS / accE.
// C layout: VGPR r, lane (g,t) -> element (M = 8*g + r, N = t).
// =====================================================================
__global__ void __launch_bounds__(256)
cc_main_kernel(const float* __restrict__ emb,
               const float* __restrict__ W,
               const float* __restrict__ rnA,
               const float* __restrict__ rnW,
               const v16h* __restrict__ bfrag,
               float* __restrict__ out) {
    const int wave = blockIdx.x * (blockDim.x >> 5) + (threadIdx.x >> 5);
    const int lane = threadIdx.x & 31;
    const int tn = wave % NTILES;
    const int tm = wave / NTILES;
    const int t = lane & 15, g = lane >> 4;

    const int Mbase = tm * 16;
    const int arow  = Mbase + t;              // A-matrix row held by this lane
    const int n     = tn * 16 + t;            // output column for this lane
    const int nclamp = (n < NCLS) ? n : (NCLS - 1);

    const float* aptr = emb + (size_t)arow * EDIM;
    const float* pang = W + (size_t)nclamp * EDIM + FEAT;   // class angle base
    const v16h*  bfb  = bfrag + (size_t)tn * JNT * 4 * 32;

    v8f accS, accE;
#pragma unroll
    for (int r = 0; r < 8; ++r) { accS[r] = 0.f; accE[r] = 0.f; }

    for (int j = 0; j < JNT; ++j) {
        // ---- per-joint 16x16x128 GEMM on tensor cores (raw values) ----
        v8f c;
#pragma unroll
        for (int r = 0; r < 8; ++r) c[r] = 0.f;

#pragma unroll
        for (int kb = 0; kb < 4; ++kb) {
            const float* ap = aptr + j * DIM + kb * 32 + g * 8;
            v16h a;
#pragma unroll
            for (int e = 0; e < 8; ++e) a[e]     = (_Float16)ap[e];
#pragma unroll
            for (int e = 0; e < 8; ++e) a[8 + e] = (_Float16)ap[16 + e];
            const v16h b = bfb[(j * 4 + kb) * 32 + lane];
            c = __builtin_amdgcn_wmma_f32_16x16x32_f16(
                    false, a, false, b, (short)0, c, false, false);
        }

        // ---- fold joint-j softmax weight into running sums ----
        const float pa0 = pang[j * ANGD + 0];
        const float pa1 = pang[j * ANGD + 1];
        const float pa2 = pang[j * ANGD + 2];
#pragma unroll
        for (int r = 0; r < 8; ++r) {
            const int m = Mbase + 8 * g + r;
            const float* xa = emb + (size_t)m * EDIM + FEAT + j * ANGD;
            const float dx = xa[0] - pa0;
            const float dy = xa[1] - pa1;
            const float dz = xa[2] - pa2;
            const float e = __expf(sqrtf(dx * dx + dy * dy + dz * dz) * TEMP_INV);
            accS[r] += e * c[r];
            accE[r] += e;
        }
    }

    if (n < NCLS) {
        const float rwn = rnW[nclamp];
#pragma unroll
        for (int r = 0; r < 8; ++r) {
            const int m = Mbase + 8 * g + r;
            const float scale = OUTSCALE * rnA[m] * rwn;
            out[(size_t)m * NCLS + n] = scale * accS[r] / accE[r];
        }
    }
}

// =====================================================================
extern "C" void kernel_launch(void* const* d_in, const int* in_sizes, int n_in,
                              void* d_out, int out_size, void* d_ws, size_t ws_size,
                              hipStream_t stream) {
    (void)in_sizes; (void)n_in; (void)out_size; (void)ws_size;

    const float* emb = (const float*)d_in[0];   // (16384, 1965) f32
    const float* W   = (const float*)d_in[1];   // (68, 1965)    f32
    float* out = (float*)d_out;                 // (16384, 68)   f32

    char* ws = (char*)d_ws;
    float* rnA   = (float*)(ws + RNORM_OFF);
    float* rnW   = (float*)(ws + WNORM_OFF);
    v16h*  bfrag = (v16h*)(ws + BFRAG_OFF);

    // 1) inverse feature norms (epilogue scale factors only)
    cc_rownorm_kernel<<<BROWS, 128, 0, stream>>>(emb, rnA);
    cc_rownorm_kernel<<<NCLS, 128, 0, stream>>>(W, rnW);

    // 2) pack raw W into f16 WMMA B fragments (300 fragments)
    cc_packB_kernel<<<NTILES * JNT * 4, 32, 0, stream>>>(W, bfrag);

    // 3) fused GEMM + softmax-weighted epilogue
    // 1024 M-tiles * 5 N-tiles = 5120 waves; 8 waves/block -> 640 blocks
    cc_main_kernel<<<640, 256, 0, stream>>>(emb, W, rnA, rnW, bfrag, out);
}
